// GNN_6708738916464
// MI455X (gfx1250) — compile-verified, matrix operand order checked by reading, not run on hardware
//
#include <hip/hip_runtime.h>
#include <hip/hip_fp16.h>

typedef __attribute__((ext_vector_type(16))) _Float16 v16h;
typedef __attribute__((ext_vector_type(8)))  float    v8f;
typedef __attribute__((ext_vector_type(8)))  _Float16 h8;
typedef __attribute__((ext_vector_type(4)))  _Float16 h4;

constexpr int   kN    = 32768;     // nodes
constexpr int   kE    = 524288;    // edges
constexpr int   kB    = 128;       // graphs
constexpr int   kNPG  = 256;       // nodes per graph
constexpr int   kHC   = 128;       // heads * channels
constexpr int   kEMB  = 64;
constexpr int   kNH   = 2;
constexpr int   kEDIM = 16;
constexpr float kNEG  = -1e30f;

// ---------------------------------------------------------------------------
// Weight pre-pack: W [K,NO] fp32 -> fp16 in WMMA B-fragment order
// layout [nt][kt][lane][j]: n = nt*16 + lane%16,
//   kl = (j/8)*16 + (lane/16)*8 + (j%8), kg = kt*32 + kl (zero-padded past K)
// ---------------------------------------------------------------------------
__global__ void pack_weights(const float* __restrict__ W, __half* __restrict__ Wp,
                             int K, int NO)
{
    int idx = blockIdx.x * blockDim.x + threadIdx.x;
    int ktiles = (K + 31) >> 5;
    int total = (NO >> 4) * ktiles * 512;
    if (idx >= total) return;
    int j    = idx & 15;
    int lane = (idx >> 4) & 31;
    int rest = idx >> 9;
    int kt   = rest % ktiles;
    int nt   = rest / ktiles;
    int n  = nt * 16 + (lane & 15);
    int kl = ((j >> 3) << 4) + ((lane >> 4) << 3) + (j & 7);
    int kg = kt * 32 + kl;
    float v = (kg < K) ? W[(size_t)kg * NO + n] : 0.0f;
    Wp[idx] = __float2half(v);
}

// fp32 -> fp16 bulk convert (4 elements / thread)
__global__ void cvt_f32_f16(const float* __restrict__ x, __half* __restrict__ y, size_t n4)
{
    size_t i = (size_t)blockIdx.x * blockDim.x + threadIdx.x;
    if (i >= n4) return;
    float4 v = ((const float4*)x)[i];
    h4 h;
    h[0] = (_Float16)v.x; h[1] = (_Float16)v.y;
    h[2] = (_Float16)v.z; h[3] = (_Float16)v.w;
    ((h4*)y)[i] = h;
}

// immediate-operand s_wait_asynccnt (constant-folded under full unroll)
__device__ __forceinline__ void wait_async_n(int n)
{
    switch (n) {
    case 0: asm volatile("s_wait_asynccnt 0x0" ::: "memory"); break;
    case 1: asm volatile("s_wait_asynccnt 0x1" ::: "memory"); break;
    case 2: asm volatile("s_wait_asynccnt 0x2" ::: "memory"); break;
    case 3: asm volatile("s_wait_asynccnt 0x3" ::: "memory"); break;
    case 4: asm volatile("s_wait_asynccnt 0x4" ::: "memory"); break;
    case 5: asm volatile("s_wait_asynccnt 0x5" ::: "memory"); break;
    case 6: asm volatile("s_wait_asynccnt 0x6" ::: "memory"); break;
    default: asm volatile("s_wait_asynccnt 0x7" ::: "memory"); break;
    }
}

// ---------------------------------------------------------------------------
// WMMA GEMM: Y[M,NO] = Xh[M,K](fp16) @ W[K,NO] (+bias), K/NO compile-time.
// grid.x = M/16, grid.y = NO/64; block = 128 (4 waves, one 16-col tile each).
// A tiles copied global->LDS with async-LDS DMA (all KT tiles issued up front,
// drained per-tile with s_wait_asynccnt); B read from packed fp16 weights
// (one contiguous 32B v16h load per lane per k-tile). Output stores use
// compile-time immediate offsets from one base address.
// ---------------------------------------------------------------------------
template<int K, int NO, bool BIAS, bool RELU, bool HALFOUT>
__global__ void wmma_linear(const __half* __restrict__ X, const __half* __restrict__ Wp,
                            const float* __restrict__ bias, float* __restrict__ Y)
{
    constexpr int KT = (K + 31) / 32;
    __shared__ alignas(16) _Float16 As[KT * 512];   // KT row-major 16x32 fp16 tiles
    const int lane  = threadIdx.x & 31;
    const int wave  = threadIdx.x >> 5;
    const int row0  = blockIdx.x * 16;
    const int nt    = blockIdx.y * 4 + wave;
    const int mloc  = lane & 15;
    const int khalf = lane >> 4;

    // staging split: thread t copies 4 halves (8B) at (row sr, col sc)
    const int ho = threadIdx.x * 4;
    const int sr = ho >> 5;             // 0..15
    const int sc = ho & 31;             // 0,4,...,28

    // zero-fill pad columns once when K % 32 != 0 (K=16 case)
    if constexpr (K % 32 != 0) {
        if (sc >= (K % 32)) {
            h4 z = {};
            *(h4*)&As[sr * 32 + sc] = z;
        }
    }

    // issue all K-tile async copies up front
    const unsigned lbase = (unsigned)(uintptr_t)(&As[0]);
    #pragma unroll
    for (int kt = 0; kt < KT; ++kt) {
        if (K % 32 == 0 || sc < (K % 32)) {
            unsigned laddr = lbase + (unsigned)((kt * 512 + sr * 32 + sc) * 2);
            const __half* gp = X + (size_t)(row0 + sr) * K + kt * 32 + sc;
            asm volatile("global_load_async_to_lds_b64 %0, %1, off"
                         :: "v"(laddr), "v"((unsigned long long)(uintptr_t)gp)
                         : "memory");
        }
    }

    const v16h* Bp = (const v16h*)Wp + (size_t)nt * KT * 32 + lane;
    v8f acc = {};
    #pragma unroll
    for (int kt = 0; kt < KT; ++kt) {
        wait_async_n(KT - 1 - kt);     // tile kt's copies complete (in-order)
        __syncthreads();               // all 4 waves' chunks visible
        const _Float16* tile = &As[kt * 512];
        h8 alo = *(const h8*)(tile + mloc * 32 + khalf * 8);
        h8 ahi = *(const h8*)(tile + mloc * 32 + 16 + khalf * 8);
        v16h a;
        #pragma unroll
        for (int j = 0; j < 8; ++j) { a[j] = alo[j]; a[j + 8] = ahi[j]; }
        v16h b = Bp[kt * 32];
        acc = __builtin_amdgcn_wmma_f32_16x16x32_f16(
                  false, a, false, b, (short)0, acc, false, false);
    }

    const int n = nt * 16 + mloc;
    float bv = 0.0f;
    if constexpr (BIAS) bv = bias[n];
    const size_t base = (size_t)(row0 + (khalf << 3)) * NO + n;
    if constexpr (HALFOUT) {
        __half* yp = (__half*)Y + base;
        #pragma unroll
        for (int r = 0; r < 8; ++r) {
            float v = acc[r] + bv;
            if (RELU) v = fmaxf(v, 0.0f);
            yp[r * NO] = __float2half(v);
        }
    } else {
        float* yp = Y + base;
        #pragma unroll
        for (int r = 0; r < 8; ++r) {
            float v = acc[r] + bv;
            if (RELU) v = fmaxf(v, 0.0f);
            yp[r * NO] = v;
        }
    }
}

// ---------------------------------------------------------------------------
// utility fills / copies
// ---------------------------------------------------------------------------
__global__ void fill_f32(float* p, float v, size_t n) {
    size_t i = (size_t)blockIdx.x * blockDim.x + threadIdx.x;
    if (i < n) p[i] = v;
}
__global__ void fill_u32(unsigned* p, unsigned v, size_t n) {
    size_t i = (size_t)blockIdx.x * blockDim.x + threadIdx.x;
    if (i < n) p[i] = v;
}
__global__ void copy_i32(int* d, const int* s, int n) {
    int i = blockIdx.x * blockDim.x + threadIdx.x;
    if (i < n) d[i] = s[i];
}

// monotone float<->uint ordering for atomicMax on floats
__device__ inline unsigned fkey(float f) {
    unsigned u = __float_as_uint(f);
    return (u & 0x80000000u) ? ~u : (u | 0x80000000u);
}
__device__ inline float fdec(unsigned u) {
    return (u & 0x80000000u) ? __uint_as_float(u & 0x7fffffffu) : __uint_as_float(~u);
}

// ---------------------------------------------------------------------------
// Edge attention pass 1: logits + segment max (ordered-uint atomicMax)
// ---------------------------------------------------------------------------
__global__ void att_logits(const float* __restrict__ q, const float* __restrict__ k,
                           const __half* __restrict__ ep,
                           const int* __restrict__ src, const int* __restrict__ dst,
                           const int* __restrict__ emask,
                           float* __restrict__ logits, unsigned* __restrict__ mxkey)
{
    int idx = blockIdx.x * blockDim.x + threadIdx.x;
    if (idx >= kE * kNH) return;
    int e = idx >> 1, h = idx & 1;
    int s = src[e], d = dst[e];
    const float4* qp = (const float4*)(q  + (size_t)d * kHC + h * kEMB);
    const float4* kp = (const float4*)(k  + (size_t)s * kHC + h * kEMB);
    const h4*     epp = (const h4*)(ep + (size_t)e * kHC + h * kEMB);
    float acc = 0.0f;
    #pragma unroll
    for (int c = 0; c < kEMB / 4; ++c) {
        float4 qv = qp[c], kv = kp[c];
        h4 ev = epp[c];
        acc += qv.x * (kv.x + (float)ev[0]) + qv.y * (kv.y + (float)ev[1]) +
               qv.z * (kv.z + (float)ev[2]) + qv.w * (kv.w + (float)ev[3]);
    }
    float lg = emask[e] ? acc * 0.125f : kNEG;   // 1/sqrt(64)
    logits[idx] = lg;
    atomicMax(&mxkey[(size_t)d * kNH + h], fkey(lg));
}

// pass 2: pe = exp(logit - m) * emask ; segment-sum denom
__global__ void att_expsum(const float* __restrict__ logits, const unsigned* __restrict__ mxkey,
                           const int* __restrict__ dst, const int* __restrict__ emask,
                           float* __restrict__ pe, float* __restrict__ denom)
{
    int idx = blockIdx.x * blockDim.x + threadIdx.x;
    if (idx >= kE * kNH) return;
    int e = idx >> 1, h = idx & 1;
    int d = dst[e];
    float m = fdec(mxkey[(size_t)d * kNH + h]);
    float p = emask[e] ? __expf(logits[idx] - m) : 0.0f;
    pe[idx] = p;
    if (p != 0.0f) atomicAdd(&denom[(size_t)d * kNH + h], p);
}

// pass 3: out[dst] += alpha * (v[src] + e)  (4 channels per thread)
__global__ void att_aggregate(const float* __restrict__ v, const __half* __restrict__ ep,
                              const float* __restrict__ pe, const float* __restrict__ denom,
                              const int* __restrict__ src, const int* __restrict__ dst,
                              float* __restrict__ attout)
{
    int idx = blockIdx.x * blockDim.x + threadIdx.x;
    if (idx >= kE * kNH * 16) return;
    int c4 = idx & 15;
    int h  = (idx >> 4) & 1;
    int e  = idx >> 5;
    float p = pe[(size_t)e * kNH + h];
    if (p == 0.0f) return;
    int s = src[e], d = dst[e];
    float w = p / fmaxf(denom[(size_t)d * kNH + h], 1e-16f);
    int base = h * kEMB + c4 * 4;
    float4 vv = *(const float4*)(v  + (size_t)s * kHC + base);
    h4     ev = *(const h4*)(ep + (size_t)e * kHC + base);
    float* op = attout + (size_t)d * kHC + base;
    atomicAdd(&op[0], w * (vv.x + (float)ev[0]));
    atomicAdd(&op[1], w * (vv.y + (float)ev[1]));
    atomicAdd(&op[2], w * (vv.z + (float)ev[2]));
    atomicAdd(&op[3], w * (vv.w + (float)ev[3]));
}

// beta gate: outh = beta*skip + (1-beta)*attout (fp16, feeds transf GEMM)
__global__ void beta_gate(const float* __restrict__ skip, const float* __restrict__ att,
                          const float* __restrict__ Wb, __half* __restrict__ outh)
{
    int n = blockIdx.x * blockDim.x + threadIdx.x;
    if (n >= kN) return;
    const float4* sp = (const float4*)(skip + (size_t)n * kHC);
    const float4* op = (const float4*)(att  + (size_t)n * kHC);
    const float4* w0 = (const float4*)(Wb);
    const float4* w1 = (const float4*)(Wb + kHC);
    const float4* w2 = (const float4*)(Wb + 2 * kHC);
    float acc = 0.0f;
    #pragma unroll
    for (int j = 0; j < kHC / 4; ++j) {
        float4 s = sp[j], o = op[j], a = w0[j], b = w1[j], c = w2[j];
        acc += a.x * s.x + b.x * o.x + c.x * (s.x - o.x);
        acc += a.y * s.y + b.y * o.y + c.y * (s.y - o.y);
        acc += a.z * s.z + b.z * o.z + c.z * (s.z - o.z);
        acc += a.w * s.w + b.w * o.w + c.w * (s.w - o.w);
    }
    float beta = 1.0f / (1.0f + __expf(-acc));
    h4* ob = (h4*)(outh + (size_t)n * kHC);
    #pragma unroll
    for (int j = 0; j < kHC / 4; ++j) {
        float4 s = sp[j], o = op[j];
        h4 r;
        r[0] = (_Float16)(beta * s.x + (1.0f - beta) * o.x);
        r[1] = (_Float16)(beta * s.y + (1.0f - beta) * o.y);
        r[2] = (_Float16)(beta * s.z + (1.0f - beta) * o.z);
        r[3] = (_Float16)(beta * s.w + (1.0f - beta) * o.w);
        ob[j] = r;
    }
}

// ---------------------------------------------------------------------------
// masked BatchNorm: stats = [sum(64), sumsq(64), cnt]
// ---------------------------------------------------------------------------
__global__ void bn_reduce(const float* __restrict__ x, const int* __restrict__ nmask,
                          float* __restrict__ stats)
{
    __shared__ float ssum[kEMB];
    __shared__ float ssq[kEMB];
    __shared__ float scnt;
    if (threadIdx.x < kEMB) { ssum[threadIdx.x] = 0.0f; ssq[threadIdx.x] = 0.0f; }
    if (threadIdx.x == 0) scnt = 0.0f;
    __syncthreads();
    size_t total = (size_t)kN * kEMB;
    for (size_t i = (size_t)blockIdx.x * blockDim.x + threadIdx.x; i < total;
         i += (size_t)gridDim.x * blockDim.x) {
        int n = (int)(i >> 6), c = (int)(i & 63);
        float w = nmask[n] ? 1.0f : 0.0f;
        float xv = x[i];
        float v = xv * w;
        atomicAdd(&ssum[c], v);
        atomicAdd(&ssq[c], v * xv);
        if (c == 0) atomicAdd(&scnt, w);
    }
    __syncthreads();
    if (threadIdx.x < kEMB) {
        atomicAdd(&stats[threadIdx.x], ssum[threadIdx.x]);
        atomicAdd(&stats[kEMB + threadIdx.x], ssq[threadIdx.x]);
    }
    if (threadIdx.x == 0) atomicAdd(&stats[2 * kEMB], scnt);
}

// also maintains the fp16 mirror used as next conv's GEMM input
__global__ void bn_apply(float* __restrict__ x, __half* __restrict__ xh,
                         const float* __restrict__ g, const float* __restrict__ b,
                         const float* __restrict__ stats, const int* __restrict__ nmask)
{
    size_t i = (size_t)blockIdx.x * blockDim.x + threadIdx.x;
    if (i >= (size_t)kN * kEMB) return;
    int n = (int)(i >> 6), c = (int)(i & 63);
    float cnt = fmaxf(stats[2 * kEMB], 1.0f);
    float mu  = stats[c] / cnt;
    float var = stats[kEMB + c] / cnt - mu * mu;
    float w = nmask[n] ? 1.0f : 0.0f;
    float r = (g[c] * (x[i] - mu) * rsqrtf(var + 1e-5f) + b[c]) * w;
    x[i]  = r;
    xh[i] = __float2half(r);
}

// ---------------------------------------------------------------------------
// top-k pooling: block per graph, bitonic sort of 256 scores in LDS
// ---------------------------------------------------------------------------
__global__ void topk_pool(float* __restrict__ x, __half* __restrict__ xh,
                          const float* __restrict__ p,
                          const int* __restrict__ nmask, int* __restrict__ nmnew, int kkeep)
{
    __shared__ float skey[kNPG];
    __shared__ int   sidx[kNPG];
    __shared__ int   flag[kNPG];
    int g = blockIdx.x, t = threadIdx.x;
    int n = g * kNPG + t;

    const float4* xv = (const float4*)(x + (size_t)n * kEMB);
    const float4* pv = (const float4*)p;
    float pn = 0.0f, sc = 0.0f;
    #pragma unroll
    for (int c = 0; c < kEMB / 4; ++c) {
        float4 a = xv[c], b = pv[c];
        pn += b.x * b.x + b.y * b.y + b.z * b.z + b.w * b.w;
        sc += a.x * b.x + a.y * b.y + a.z * b.z + a.w * b.w;
    }
    sc *= rsqrtf(pn);
    float scm = nmask[n] ? sc : kNEG;

    skey[t] = scm; sidx[t] = t; flag[t] = 0;
    __syncthreads();

    // bitonic sort, descending (ties: smaller index ranks higher, like top_k)
    for (int size = 2; size <= kNPG; size <<= 1) {
        for (int stride = size >> 1; stride > 0; stride >>= 1) {
            int ixj = t ^ stride;
            if (ixj > t) {
                bool dirDesc = ((t & size) == 0);
                float a = skey[t], bk = skey[ixj];
                int ia = sidx[t], ib = sidx[ixj];
                bool aLess = (a < bk) || (a == bk && ia > ib);
                if (dirDesc ? aLess : !aLess) {
                    skey[t] = bk; skey[ixj] = a; sidx[t] = ib; sidx[ixj] = ia;
                }
            }
            __syncthreads();
        }
    }
    if (t < kkeep) flag[sidx[t]] = 1;
    __syncthreads();

    int keep = flag[t];
    nmnew[n] = keep;
    float fac = tanhf(scm) * (keep ? 1.0f : 0.0f);
    float4* xw = (float4*)(x + (size_t)n * kEMB);
    h4*     hw = (h4*)(xh + (size_t)n * kEMB);
    #pragma unroll
    for (int c = 0; c < kEMB / 4; ++c) {
        float4 a = xw[c];
        a.x *= fac; a.y *= fac; a.z *= fac; a.w *= fac;
        xw[c] = a;
        h4 hv;
        hv[0] = (_Float16)a.x; hv[1] = (_Float16)a.y;
        hv[2] = (_Float16)a.z; hv[3] = (_Float16)a.w;
        hw[c] = hv;
    }
}

__global__ void update_emask(int* __restrict__ emask, const int* __restrict__ nm,
                             const int* __restrict__ src, const int* __restrict__ dst)
{
    int e = blockIdx.x * blockDim.x + threadIdx.x;
    if (e >= kE) return;
    emask[e] = emask[e] && nm[src[e]] && nm[dst[e]];
}

// per-graph masked max/mean readout; accumulates into g[b, 0:64]=max, [64:128]=mean
__global__ void readout(const float* __restrict__ x, const int* __restrict__ nmask,
                        float* __restrict__ gacc)
{
    int b = blockIdx.x, c = threadIdx.x;   // blockDim = 64
    float mx = kNEG, sm = 0.0f, cnt = 0.0f;
    for (int t = 0; t < kNPG; ++t) {
        int n = b * kNPG + t;
        if (nmask[n]) {
            float v = x[(size_t)n * kEMB + c];
            mx = fmaxf(mx, v); sm += v; cnt += 1.0f;
        }
    }
    gacc[b * 128 + c]      += mx;
    gacc[b * 128 + 64 + c] += sm / fmaxf(cnt, 1.0f);
}

// final linear [128]->1
__global__ void lin_out(const float* __restrict__ h, const float* __restrict__ W,
                        const float* __restrict__ b, float* __restrict__ out)
{
    int n = blockIdx.x * blockDim.x + threadIdx.x;
    if (n >= kB) return;
    const float4* hp = (const float4*)(h + (size_t)n * 128);
    const float4* wp = (const float4*)W;
    float acc = b[0];
    #pragma unroll
    for (int j = 0; j < 32; ++j) {
        float4 a = hp[j], ww = wp[j];
        acc += a.x * ww.x + a.y * ww.y + a.z * ww.z + a.w * ww.w;
    }
    out[n] = acc;
}

// ---------------------------------------------------------------------------
static inline unsigned host_fkey(float f) {
    unsigned u; __builtin_memcpy(&u, &f, 4);
    return (u & 0x80000000u) ? ~u : (u | 0x80000000u);
}
static inline int cdiv(long long a, long long b) { return (int)((a + b - 1) / b); }

extern "C" void kernel_launch(void* const* d_in, const int* in_sizes, int n_in,
                              void* d_out, int out_size, void* d_ws, size_t ws_size,
                              hipStream_t stream)
{
    const float* x_in  = (const float*)d_in[0];
    const float* eattr = (const float*)d_in[1];
    const int*   eidx  = (const int*)d_in[2];
    const int* src = eidx;
    const int* dst = eidx + kE;
    auto P = [&](int i) { return (const float*)d_in[i]; };
    const int base_convs = 4, base_transf = 54, base_bn = 64, base_pool = 74, base_head = 76;

    // workspace carve (floats, 32B-aligned chunks)
    float* w = (float*)d_ws;
    size_t o = 0;
    auto alloc = [&](size_t n) { float* p = w + o; o += (n + 7) & ~(size_t)7; return p; };
    float*    xf      = alloc((size_t)kN * kEMB);
    float*    qb      = alloc((size_t)kN * kHC);
    float*    kb      = alloc((size_t)kN * kHC);
    float*    vb      = alloc((size_t)kN * kHC);
    float*    skipb   = alloc((size_t)kN * kHC);
    float*    attoutb = alloc((size_t)kN * kHC);
    __half*   eprojb  = (__half*)alloc((size_t)kE * kHC / 2);
    float*    logitsb = alloc((size_t)kE * kNH);
    float*    peb     = alloc((size_t)kE * kNH);
    unsigned* mxk     = (unsigned*)alloc((size_t)kN * kNH);
    float*    denomb  = alloc((size_t)kN * kNH);
    int*      nmaskb  = (int*)alloc(kN);
    int*      nmnewb  = (int*)alloc(kN);
    int*      emaskb  = (int*)alloc(kE);
    float*    gaccb   = alloc(kB * 128);
    float*    statsb  = alloc(160);
    float*    h2      = alloc((size_t)kB * 128);
    // fp16 activation buffers (GEMM inputs)
    __half*   xh0     = (__half*)alloc((size_t)kN * 32 / 2);    // x_in fp16
    __half*   eah     = (__half*)alloc((size_t)kE * kEDIM / 2); // edge_attr fp16
    __half*   convh   = (__half*)alloc((size_t)kN * kHC / 2);   // conv output fp16
    __half*   xfh     = (__half*)alloc((size_t)kN * kEMB / 2);  // xf fp16 mirror
    __half*   gah     = (__half*)alloc((size_t)kB * 128 / 2);
    __half*   h1h     = (__half*)alloc((size_t)kB * 256 / 2);
    __half*   packArena = (__half*)alloc(512 * 1024 / 2);       // packed fp16 weights

    // ---- pack all weights into WMMA B-fragment order (fp16) ----
    size_t po = 0;
    auto packW = [&](const float* Wsrc, int K, int NO) {
        int ktiles = (K + 31) >> 5;
        size_t tot = (size_t)(NO >> 4) * ktiles * 512;
        __half* dstp = packArena + po;
        po += (tot + 15) & ~(size_t)15;
        pack_weights<<<cdiv(tot, 256), 256, 0, stream>>>(Wsrc, dstp, K, NO);
        return dstp;
    };
    __half *pWq[5], *pWk[5], *pWv[5], *pWs[5], *pWe[5], *pWt[5];
    for (int ci = 0; ci < 5; ++ci) {
        int b0 = base_convs + ci * 10;
        int fi = ci ? kEMB : 32;
        pWq[ci] = packW(P(b0 + 0), fi,    kHC);
        pWk[ci] = packW(P(b0 + 2), fi,    kHC);
        pWv[ci] = packW(P(b0 + 4), fi,    kHC);
        pWs[ci] = packW(P(b0 + 7), fi,    kHC);
        pWe[ci] = packW(P(b0 + 6), kEDIM, kHC);
        pWt[ci] = packW(P(base_transf + ci * 2), kHC, kEMB);
    }
    __half* pL1 = packW(P(base_head + 0), 128, 256);
    __half* pL2 = packW(P(base_head + 2), 256, 128);

    // fp16 copies of the raw inputs
    cvt_f32_f16<<<cdiv((size_t)kN * 32 / 4, 256), 256, 0, stream>>>(x_in, xh0, (size_t)kN * 32 / 4);
    cvt_f32_f16<<<cdiv((size_t)kE * kEDIM / 4, 256), 256, 0, stream>>>(eattr, eah, (size_t)kE * kEDIM / 4);

    fill_u32<<<cdiv(kN, 256), 256, 0, stream>>>((unsigned*)nmaskb, 1u, kN);
    fill_u32<<<cdiv(kE, 256), 256, 0, stream>>>((unsigned*)emaskb, 1u, kE);
    fill_f32<<<cdiv(kB * 128, 256), 256, 0, stream>>>(gaccb, 0.0f, kB * 128);

    auto conv = [&](const __half* xin, int fi, int ci) {
        int b0 = base_convs + ci * 10;
        dim3 gN(kN / 16, kHC / 64);
        dim3 gE(kE / 16, kHC / 64);
        if (fi == 32) {
            wmma_linear<32, 128, true, false, false><<<gN, 128, 0, stream>>>(xin, pWq[ci], P(b0 + 1), qb);
            wmma_linear<32, 128, true, false, false><<<gN, 128, 0, stream>>>(xin, pWk[ci], P(b0 + 3), kb);
            wmma_linear<32, 128, true, false, false><<<gN, 128, 0, stream>>>(xin, pWv[ci], P(b0 + 5), vb);
            wmma_linear<32, 128, true, false, false><<<gN, 128, 0, stream>>>(xin, pWs[ci], P(b0 + 8), skipb);
        } else {
            wmma_linear<64, 128, true, false, false><<<gN, 128, 0, stream>>>(xin, pWq[ci], P(b0 + 1), qb);
            wmma_linear<64, 128, true, false, false><<<gN, 128, 0, stream>>>(xin, pWk[ci], P(b0 + 3), kb);
            wmma_linear<64, 128, true, false, false><<<gN, 128, 0, stream>>>(xin, pWv[ci], P(b0 + 5), vb);
            wmma_linear<64, 128, true, false, false><<<gN, 128, 0, stream>>>(xin, pWs[ci], P(b0 + 8), skipb);
        }
        wmma_linear<16, 128, false, false, true><<<gE, 128, 0, stream>>>(eah, pWe[ci], nullptr, (float*)eprojb);

        fill_u32<<<cdiv((size_t)kN * kNH, 256), 256, 0, stream>>>(mxk, host_fkey(kNEG), (size_t)kN * kNH);
        fill_f32<<<cdiv((size_t)kN * kNH, 256), 256, 0, stream>>>(denomb, 0.0f, (size_t)kN * kNH);
        fill_f32<<<cdiv((size_t)kN * kHC, 256), 256, 0, stream>>>(attoutb, 0.0f, (size_t)kN * kHC);

        att_logits<<<cdiv((size_t)kE * kNH, 256), 256, 0, stream>>>(
            qb, kb, eprojb, src, dst, emaskb, logitsb, mxk);
        att_expsum<<<cdiv((size_t)kE * kNH, 256), 256, 0, stream>>>(
            logitsb, mxk, dst, emaskb, peb, denomb);
        att_aggregate<<<cdiv((size_t)kE * kNH * 16, 256), 256, 0, stream>>>(
            vb, eprojb, peb, denomb, src, dst, attoutb);
        beta_gate<<<cdiv(kN, 256), 256, 0, stream>>>(skipb, attoutb, P(b0 + 9), convh);
    };

    auto transf_bn = [&](int ti) {
        dim3 gT(kN / 16, kEMB / 64);
        wmma_linear<128, 64, true, true, false><<<gT, 128, 0, stream>>>(
            convh, pWt[ti], P(base_transf + ti * 2 + 1), xf);
        fill_f32<<<1, 256, 0, stream>>>(statsb, 0.0f, 160);
        bn_reduce<<<512, 256, 0, stream>>>(xf, nmaskb, statsb);
        bn_apply<<<cdiv((size_t)kN * kEMB, 256), 256, 0, stream>>>(
            xf, xfh, P(base_bn + ti * 2), P(base_bn + ti * 2 + 1), statsb, nmaskb);
    };

    conv(xh0, 32, 0);
    transf_bn(0);

    int valid = kNPG, pool_i = 0;
    for (int i = 0; i < 4; ++i) {
        conv(xfh, kEMB, i + 1);
        transf_bn(i + 1);
        if (i % 2 == 0) {
            valid = (valid + 1) / 2;   // ceil(0.5 * valid): 128 then 64
            topk_pool<<<kB, kNPG, 0, stream>>>(xf, xfh, P(base_pool + pool_i), nmaskb, nmnewb, valid);
            update_emask<<<cdiv(kE, 256), 256, 0, stream>>>(emaskb, nmnewb, src, dst);
            copy_i32<<<cdiv(kN, 256), 256, 0, stream>>>(nmaskb, nmnewb, kN);
            readout<<<kB, kEMB, 0, stream>>>(xf, nmaskb, gaccb);
            ++pool_i;
        }
    }

    // MLP head: 128 -> 256 -> 128 -> 1
    cvt_f32_f16<<<cdiv(kB * 128 / 4, 256), 256, 0, stream>>>(gaccb, gah, kB * 128 / 4);
    wmma_linear<128, 256, true, true, true><<<dim3(kB / 16, 256 / 64), 128, 0, stream>>>(
        gah, pL1, P(base_head + 1), (float*)h1h);
    wmma_linear<256, 128, true, true, false><<<dim3(kB / 16, 128 / 64), 128, 0, stream>>>(
        h1h, pL2, P(base_head + 3), h2);
    lin_out<<<1, kB, 0, stream>>>(h2, P(base_head + 4), P(base_head + 5), (float*)d_out);
}
